// TiedCirculantSpectralFilter_74955769249888
// MI455X (gfx1250) — matrix operand order repaired and checked by reading, not compiled, not run on hardware
//
#include <hip/hip_runtime.h>
#include <hip/hip_bf16.h>

// Problem constants (from reference): B=512, D=1024, P=64, K=16, L_S=16
#define BSZ 512
#define DD  1024
#define PP  64
#define KK  16
#define PK  80      // P + K
#define LS  16

#define KC   128            // K-chunk per pipeline stage
#define KCP  136            // padded row length (272 B: 16B-aligned, bank-skewed)
#define NCH  (DD / KC)      // 8 chunks

typedef __attribute__((ext_vector_type(16))) __bf16 v16bf;
typedef __attribute__((ext_vector_type(8)))  __bf16 v8bf;
typedef __attribute__((ext_vector_type(8)))  float  v8f;

// ---- LDS layout (bytes), double-buffered chunks -------------------------
#define WT0_OFF  0                              // [80][136] bf16 = 21760
#define XT0_OFF  (WT0_OFF + PK * KCP * 2)       // [64][136] bf16 = 17408
#define WT1_OFF  (XT0_OFF + PP * KCP * 2)       // 39168
#define XT1_OFF  (WT1_OFF + PK * KCP * 2)       // 60928
#define S_OFF    (XT1_OFF + PP * KCP * 2)       // 78336 : [80][64] f32
#define G_OFF    (S_OFF  + PK * PP * 4)         // 98816 : [1024] f32
#define H_OFF    (G_OFF  + DD * 4)              // 102912: [80] f32 (+pad)
#define SMEM_BYTES (H_OFF + 128 * 4)            // 103424 B -> up to 3 blocks/WGP

#define NTHREADS 320   // 10 wave32 waves: 5 M-tiles x 2 waves, 2 N-tiles each

__device__ __forceinline__ unsigned short f2bfbits(float f) {
    // round-to-nearest-even f32 -> bf16 bit pattern
    unsigned int u = __float_as_uint(f);
    u += 0x7FFFu + ((u >> 16) & 1u);
    return (unsigned short)(u >> 16);
}

union Pack8 { unsigned short s[8]; uint4 u; };

__global__ __launch_bounds__(NTHREADS, 2)
void tied_filter_kernel(const float* __restrict__ X,      // [B, D, 80]
                        const float* __restrict__ y,      // [B, 64]
                        const float* __restrict__ gamma,  // [D]
                        float* __restrict__ out)          // [B, 16]
{
    extern __shared__ __align__(128) char smem[];
    float* S  = (float*)(smem + S_OFF);
    float* gS = (float*)(smem + G_OFF);
    float* hS = (float*)(smem + H_OFF);

    const int b   = blockIdx.x;
    const int tid = threadIdx.x;

    // ---- stage gamma into LDS ------------------------------------------
    for (int i = tid; i < DD; i += NTHREADS) gS[i] = gamma[i];

    // ---- loader mapping: thread owns row m, 4 groups of 8 consecutive d
    const int mld = tid % PK;        // 0..79
    const int tq  = tid / PK;        // 0..3
    const float* Xb = X + (size_t)b * (DD * PK);

    // ---- WMMA wave mapping ---------------------------------------------
    const int wave = tid >> 5;               // 0..9
    const int lane = tid & 31;
    const int mt   = wave >> 1;              // M tile 0..4
    const int nt0  = (wave & 1) * 2;         // N tile base 0 or 2
    const int mfr  = mt * 16 + (lane & 15);
    const int nfr0 = nt0 * 16 + (lane & 15);
    const int nfr1 = nfr0 + 16;
    const int asel = (lane >> 4) * 8;        // A lane K-select (ISA layout)
    const int bsel = (lane >> 4) * 8;        // B lane K-select (ISA layout)

    v8f acc0 = {};
    v8f acc1 = {};
    float r[4][8];                           // register-staged chunk (32 f32)

    // ---- prologue: global-load chunk 0 ----------------------------------
#pragma unroll
    for (int k = 0; k < 4; ++k) {
        const float* p = Xb + (size_t)((tq + 4 * k) * 8) * PK + mld;
#pragma unroll
        for (int j = 0; j < 8; ++j) r[k][j] = p[(size_t)j * PK];
    }
    __syncthreads();                         // gS ready

    // convert+store chunk 0 into buffer 0
    {
        unsigned short* Wb  = (unsigned short*)(smem + WT0_OFF);
        unsigned short* XtB = (unsigned short*)(smem + XT0_OFF);
#pragma unroll
        for (int k = 0; k < 4; ++k) {
            const int dl = (tq + 4 * k) * 8;
            float4 g0 = *(const float4*)(gS + dl);
            float4 g1 = *(const float4*)(gS + dl + 4);
            float gg[8] = {g0.x, g0.y, g0.z, g0.w, g1.x, g1.y, g1.z, g1.w};
            Pack8 w, xp;
#pragma unroll
            for (int j = 0; j < 8; ++j) {
                w.s[j]  = f2bfbits(gg[j] * r[k][j]);
                xp.s[j] = f2bfbits(r[k][j]);
            }
            *(uint4*)(Wb + mld * KCP + dl) = w.u;
            if (mld < PP) *(uint4*)(XtB + mld * KCP + dl) = xp.u;
        }
    }
    __syncthreads();

    // ---- software-pipelined main loop: WMMA chunk c, prefetch c+1 -------
    for (int c = 0; c < NCH; ++c) {
        const int  cur  = c & 1;
        const bool more = (c + 1 < NCH);

        // issue next chunk's global loads early (latency hidden by WMMA)
        if (more) {
#pragma unroll
            for (int k = 0; k < 4; ++k) {
                const float* p = Xb + (size_t)((c + 1) * KC + (tq + 4 * k) * 8) * PK + mld;
#pragma unroll
                for (int j = 0; j < 8; ++j) r[k][j] = p[(size_t)j * PK];
            }
        }

        // WMMA over current buffer: KC/32 = 4 k-steps, 2 tiles
        {
            const __bf16* WbC = (const __bf16*)(smem + (cur ? WT1_OFF : WT0_OFF));
            const __bf16* XtC = (const __bf16*)(smem + (cur ? XT1_OFF : XT0_OFF));
            const __bf16* Arow  = WbC + mfr * KCP;
            const __bf16* Brow0 = XtC + nfr0 * KCP;
            const __bf16* Brow1 = XtC + nfr1 * KCP;
#pragma unroll
            for (int kb = 0; kb < KC; kb += 32) {
                v16bf a, b0, b1;
                ((v8bf*)&a)[0]  = *(const v8bf*)(Arow + kb + asel);      // K: +0..7
                ((v8bf*)&a)[1]  = *(const v8bf*)(Arow + kb + asel + 16); // K: +16..23
                ((v8bf*)&b0)[0] = *(const v8bf*)(Brow0 + kb + bsel);
                ((v8bf*)&b0)[1] = *(const v8bf*)(Brow0 + kb + bsel + 16);
                ((v8bf*)&b1)[0] = *(const v8bf*)(Brow1 + kb + bsel);
                ((v8bf*)&b1)[1] = *(const v8bf*)(Brow1 + kb + bsel + 16);
                acc0 = __builtin_amdgcn_wmma_f32_16x16x32_bf16(
                    false, a, false, b0, (short)0, acc0, false, false);
                acc1 = __builtin_amdgcn_wmma_f32_16x16x32_bf16(
                    false, a, false, b1, (short)0, acc1, false, false);
            }
        }

        // convert+store next chunk into the other buffer
        if (more) {
            unsigned short* WbN = (unsigned short*)(smem + (cur ? WT0_OFF : WT1_OFF));
            unsigned short* XtN = (unsigned short*)(smem + (cur ? XT0_OFF : XT1_OFF));
#pragma unroll
            for (int k = 0; k < 4; ++k) {
                const int dl = (tq + 4 * k) * 8;
                const int dg = (c + 1) * KC + dl;
                float4 g0 = *(const float4*)(gS + dg);
                float4 g1 = *(const float4*)(gS + dg + 4);
                float gg[8] = {g0.x, g0.y, g0.z, g0.w, g1.x, g1.y, g1.z, g1.w};
                Pack8 w, xp;
#pragma unroll
                for (int j = 0; j < 8; ++j) {
                    w.s[j]  = f2bfbits(gg[j] * r[k][j]);
                    xp.s[j] = f2bfbits(r[k][j]);
                }
                *(uint4*)(WbN + mld * KCP + dl) = w.u;
                if (mld < PP) *(uint4*)(XtN + mld * KCP + dl) = xp.u;
            }
        }
        __syncthreads();   // stores of buf^1 visible; wmma reads of buf done
    }

    // ---- write S tiles to LDS -------------------------------------------
    // C/D layout: VGPR r, lanes 0-15 -> M=r, lanes 16-31 -> M=r+8; N=lane&15
    {
        const int mrow = mt * 16 + (lane >> 4) * 8;
        const int ncol = lane & 15;
#pragma unroll
        for (int rr = 0; rr < 8; ++rr) {
            S[(mrow + rr) * PP + nt0 * 16 + ncol]       = acc0[rr];
            S[(mrow + rr) * PP + (nt0 + 1) * 16 + ncol] = acc1[rr];
        }
    }
    __syncthreads();

    // ---- 16-step recurrence on h[80] (tiny; threads 0..79) --------------
    if (tid < PK) hS[tid] = (tid < PP) ? y[(size_t)b * PP + tid] : 0.0f;
    __syncthreads();

    const float scale = (1.0f / (float)LS) * (1.0f / (float)PP);  // invL / P
    for (int it = 0; it < LS; ++it) {
        float dot = 0.0f;
        if (tid < PK) {
            const float* Srow = S + tid * PP;
#pragma unroll 8
            for (int n = 0; n < PP; ++n) dot += Srow[n] * hS[n];
        }
        __syncthreads();
        if (tid < PK) {
            float sgn = (tid < PP) ? -1.0f : 1.0f;
            hS[tid] += sgn * scale * dot;
        }
        __syncthreads();
    }

    if (tid >= PP && tid < PK)
        out[(size_t)b * KK + (tid - PP)] = hS[tid];
}

extern "C" void kernel_launch(void* const* d_in, const int* in_sizes, int n_in,
                              void* d_out, int out_size, void* d_ws, size_t ws_size,
                              hipStream_t stream) {
    (void)in_sizes; (void)n_in; (void)out_size; (void)d_ws; (void)ws_size;
    const float* X     = (const float*)d_in[0];   // [512,1024,80] f32
    const float* yTr   = (const float*)d_in[1];   // [512,64] f32
    const float* gamma = (const float*)d_in[2];   // [1024] f32
    float* out         = (float*)d_out;           // [512,16] f32

    static bool attr_set = false;  // idempotent host attribute, not stream state
    if (!attr_set) {
        hipFuncSetAttribute((const void*)tied_filter_kernel,
                            hipFuncAttributeMaxDynamicSharedMemorySize,
                            SMEM_BYTES);
        attr_set = true;
    }

    tied_filter_kernel<<<dim3(BSZ), dim3(NTHREADS), SMEM_BYTES, stream>>>(
        X, yTr, gamma, out);
}
